// TransformerBlock_51754355917412
// MI455X (gfx1250) — matrix operand order repaired and checked by reading, not compiled, not run on hardware
//
#include <hip/hip_runtime.h>
#include <hip/hip_bf16.h>
#include <cstdint>
#include <cstddef>

#define DM   1024
#define NH   16
#define DKH  64
#define DFF  4096
#define SEQL 2048
#define NB   2
#define NTOK (NB*SEQL)   // 4096 tokens
#define EPSF 1e-5f

typedef __attribute__((ext_vector_type(8)))  _Float16 v8h;
typedef __attribute__((ext_vector_type(16))) _Float16 v16h;
typedef __attribute__((ext_vector_type(8)))  float    v8f;
typedef int g4i __attribute__((vector_size(16)));   // matches builtin's pointee type

__device__ __forceinline__ v16h hcat(v8h a, v8h b){
  return __builtin_shufflevector(a, b, 0,1,2,3,4,5,6,7,8,9,10,11,12,13,14,15);
}
__device__ __forceinline__ v8f wmma_f16(v16h a, v16h b, v8f c){
  // (neg_a, A, neg_b, B, c_mod, C, reuse_a, reuse_b)
  return __builtin_amdgcn_wmma_f32_16x16x32_f16(false, a, false, b, (short)0, c, false, false);
}

// ---- CDNA5 async global->LDS copy (ASYNCcnt path), with safe fallback -----
#if __has_builtin(__builtin_amdgcn_global_load_async_to_lds_b128)
#define HAVE_ASYNC 1
#else
#define HAVE_ASYNC 0
#endif

__device__ __forceinline__ void async_cp16(const _Float16* g, _Float16* l){
#if HAVE_ASYNC
  __builtin_amdgcn_global_load_async_to_lds_b128(
      (__attribute__((address_space(1))) g4i*)g,
      (__attribute__((address_space(3))) g4i*)l, 0, 0);
#else
  *(v8h*)l = *(const v8h*)g;
#endif
}
__device__ __forceinline__ void wait_async(){
#if __has_builtin(__builtin_amdgcn_s_wait_asynccnt)
  __builtin_amdgcn_s_wait_asynccnt(0);
#elif HAVE_ASYNC
  asm volatile("s_wait_asynccnt 0" ::: "memory");
#endif
}

// ---------------------------------------------------------------------------
// f32 -> f16 conversion (weights)
// ---------------------------------------------------------------------------
__global__ __launch_bounds__(256) void cvt_f32_f16(const float* __restrict__ in,
                                                   _Float16* __restrict__ out, int n){
  int i = blockIdx.x * 256 + threadIdx.x;
  if (i < n) out[i] = (_Float16)in[i];
}

// ---------------------------------------------------------------------------
// RMSNorm: one block per token row of 1024, writes f16
// ---------------------------------------------------------------------------
__global__ __launch_bounds__(256) void rmsnorm_kernel(const float* __restrict__ x,
                                                      const float* __restrict__ g,
                                                      _Float16* __restrict__ out){
  __shared__ float red[256];
  const int t = blockIdx.x, tid = threadIdx.x;
  const float* xp = x + (size_t)t * DM;
  float s = 0.f;
  for (int i = tid; i < DM; i += 256){ float v = xp[i]; s += v * v; }
  red[tid] = s; __syncthreads();
  for (int o = 128; o > 0; o >>= 1){
    if (tid < o) red[tid] += red[tid + o];
    __syncthreads();
  }
  const float rinv = rsqrtf(red[0] * (1.f / DM) + EPSF);
  _Float16* op = out + (size_t)t * DM;
  for (int i = tid; i < DM; i += 256) op[i] = (_Float16)(xp[i] * rinv * g[i]);
}

// ---------------------------------------------------------------------------
// RoPE in-place on f16 [4096][1024]; pair index = tok*512 + h*32 + j
// ---------------------------------------------------------------------------
__global__ __launch_bounds__(256) void rope_kernel(_Float16* __restrict__ t){
  const int i   = blockIdx.x * 256 + threadIdx.x;
  const int tok = i >> 9, p = i & 511;
  const int h = p >> 5, j = p & 31;
  const int s = tok & (SEQL - 1);
  const float inv = __powf(10000.f, -(float)(2 * j) * (1.f / DKH));
  float sn, c;
  __sincosf((float)s * inv, &sn, &c);
  const size_t base = (size_t)tok * DM + h * DKH + 2 * j;
  const float x1 = (float)t[base], x2 = (float)t[base + 1];
  t[base]     = (_Float16)(x1 * c - x2 * sn);
  t[base + 1] = (_Float16)(x2 * c + x1 * sn);
}

// ---------------------------------------------------------------------------
// Tiled WMMA GEMM: C[M][N] = A[M][K] * Bt[N][K]^T   (both f16, f32 accum)
// 256 thr = 8 waves -> 128x128 tile, each wave 32x64 (2x4 frags).
// Double-buffered LDS, async global->LDS staging, prefetch 2 tiles ahead.
// EPI: 0 = store f16
//      1 = store f16 into V-transposed layout vT[(b*NH+h)*DKH+d][SEQL]
//      2 = silu(acc) -> f16
//      3 = acc * aux_f16 -> f16          (SwiGLU gate multiply)
//      4 = aux_f32 + acc -> f32          (fused residual add)
// ---------------------------------------------------------------------------
template<int EPI>
__global__ __launch_bounds__(256) void gemm_wmma_kernel(
    const _Float16* __restrict__ A, const _Float16* __restrict__ Bt,
    void* __restrict__ out, const void* __restrict__ aux,
    int M, int N, int K)
{
  __shared__ _Float16 sA[2][128][40];
  __shared__ _Float16 sB[2][128][40];
  const int tid  = threadIdx.x;
  const int lane = tid & 31, wid = tid >> 5;
  const int l15  = lane & 15;
  const bool hiHalf = lane >= 16;
  const int wm = wid >> 1, wn = wid & 1;
  const int m0 = blockIdx.y * 128, n0 = blockIdx.x * 128;

  v8f acc[2][4];
  #pragma unroll
  for (int i = 0; i < 2; ++i)
    #pragma unroll
    for (int j = 0; j < 4; ++j)
      #pragma unroll
      for (int r = 0; r < 8; ++r) acc[i][j][r] = 0.f;

  const int nk = K >> 5;

  // stage one 128x32 tile of A and Bt into LDS buffer `buf`
  auto stage = [&](int kt, int buf){
    const int k0 = kt << 5;
    for (int c = tid; c < 512; c += 256){
      const int row = c >> 2, kc = (c & 3) << 3;
      async_cp16(A  + (size_t)(m0 + row) * K + k0 + kc, &sA[buf][row][kc]);
      async_cp16(Bt + (size_t)(n0 + row) * K + k0 + kc, &sB[buf][row][kc]);
    }
  };

  stage(0, 0);
  wait_async();
  __syncthreads();

  for (int kt = 0; kt < nk; ++kt){
    const int buf = kt & 1;
    if (kt + 1 < nk) stage(kt + 1, buf ^ 1);       // overlap copy with compute
    if (kt + 2 < nk){                              // L2 prefetch 2 tiles ahead
      const int kp = (kt + 2) << 5;
      __builtin_prefetch(A  + (size_t)(m0 + (tid >> 1)) * K + kp, 0, 1);
      __builtin_prefetch(Bt + (size_t)(n0 + (tid >> 1)) * K + kp, 0, 1);
    }

    v16h af[2], bf[4];
    #pragma unroll
    for (int fm = 0; fm < 2; ++fm){
      // A frag (16x32): lane<16 -> K {0..7,16..23}; lane>=16 -> K {8..15,24..31}
      const _Float16* ap = &sA[buf][wm * 32 + fm * 16 + l15][hiHalf ? 8 : 0];
      af[fm] = hcat(*(const v8h*)ap, *(const v8h*)(ap + 16));
    }
    #pragma unroll
    for (int fn = 0; fn < 4; ++fn){
      // B frag (32x16): lane<16 -> K {0..7,16..23} of col l15; lane>=16 -> K {8..15,24..31}
      const _Float16* bp = &sB[buf][wn * 64 + fn * 16 + l15][hiHalf ? 8 : 0];
      bf[fn] = hcat(*(const v8h*)bp, *(const v8h*)(bp + 16));
    }
    #pragma unroll
    for (int fm = 0; fm < 2; ++fm)
      #pragma unroll
      for (int fn = 0; fn < 4; ++fn)
        acc[fm][fn] = wmma_f16(af[fm], bf[fn], acc[fm][fn]);

    wait_async();        // next buffer fully landed in LDS
    __syncthreads();     // all waves done reading current buffer
  }

  // epilogue: C/D layout: M = r + 8*(lane>=16), N = lane&15
  #pragma unroll
  for (int fm = 0; fm < 2; ++fm){
    #pragma unroll
    for (int fn = 0; fn < 4; ++fn){
      #pragma unroll
      for (int r = 0; r < 8; ++r){
        const float val = acc[fm][fn][r];
        const int row = m0 + wm * 32 + fm * 16 + r + (hiHalf ? 8 : 0);
        const int col = n0 + wn * 64 + fn * 16 + l15;
        const size_t idx = (size_t)row * N + col;
        if (EPI == 0){
          ((_Float16*)out)[idx] = (_Float16)val;
        } else if (EPI == 1){
          const int b = row >> 11, s = row & (SEQL - 1);
          const int hh = col >> 6, d = col & 63;
          ((_Float16*)out)[(((size_t)(b * NH + hh) * DKH + d) << 11) + s] = (_Float16)val;
        } else if (EPI == 2){
          const float sg = 1.f / (1.f + __expf(-val));
          ((_Float16*)out)[idx] = (_Float16)(val * sg);
        } else if (EPI == 3){
          const float mu = (float)((const _Float16*)aux)[idx];
          ((_Float16*)out)[idx] = (_Float16)(val * mu);
        } else {
          ((float*)out)[idx] = ((const float*)aux)[idx] + val;
        }
      }
    }
  }
}

// ---------------------------------------------------------------------------
// Flash-style causal attention. One wave per 16-query tile.
// q,k: f16 [tok][h*64+d] (RoPE'd). vT: f16 [(b*NH+h)*DKH + d][SEQL].
// ctx out: f16 [tok][h*64+d].
// Row-sums of P are computed on the matrix core via a ones-column B fragment
// (l = P x 1), removing the sum shuffle-reduction entirely.
// ---------------------------------------------------------------------------
__global__ __launch_bounds__(256) void flash_attn_kernel(
    const _Float16* __restrict__ q, const _Float16* __restrict__ k,
    const _Float16* __restrict__ vT, _Float16* __restrict__ ctx)
{
  __shared__ _Float16 pbuf[8][16][40];
  const int tid = threadIdx.x, lane = tid & 31, wid = tid >> 5;
  const int l15 = lane & 15;
  const bool hiHalf = lane >= 16;
  const int wg = blockIdx.x * 8 + wid;       // 0..4095
  const int qt = wg & 127, bh = wg >> 7;     // 128 q-tiles per (b,h)
  const int b = bh >> 4, h = bh & 15;
  const int q0 = qt << 4;

  // Q tile 16x64 -> two A fragments (K = 0..31, 32..63)
  v16h qa[2];
  {
    const _Float16* qrow = q + (size_t)(b * SEQL + q0 + l15) * DM + h * DKH;
    #pragma unroll
    for (int kk = 0; kk < 2; ++kk){
      const _Float16* p = qrow + kk * 32 + (hiHalf ? 8 : 0);
      qa[kk] = hcat(*(const v8h*)p, *(const v8h*)(p + 16));
    }
  }

  v16h ones;
  #pragma unroll
  for (int i = 0; i < 16; ++i) ones[i] = (_Float16)1.f;

  v8f o[4], lacc;
  float m[8];
  #pragma unroll
  for (int nt = 0; nt < 4; ++nt)
    #pragma unroll
    for (int r = 0; r < 8; ++r) o[nt][r] = 0.f;
  #pragma unroll
  for (int r = 0; r < 8; ++r){ m[r] = -1e30f; lacc[r] = 0.f; }

  for (int k0 = 0; k0 <= q0; k0 += 32){
    // scores: two 16x16 C tiles over 32 keys
    v8f sc[2];
    #pragma unroll
    for (int ct = 0; ct < 2; ++ct){
      v8f a;
      #pragma unroll
      for (int r = 0; r < 8; ++r) a[r] = 0.f;
      const int key = k0 + ct * 16 + l15;
      const _Float16* krow = k + (size_t)(b * SEQL + key) * DM + h * DKH;
      #pragma unroll
      for (int kk = 0; kk < 2; ++kk){
        const _Float16* p2 = krow + kk * 32 + (hiHalf ? 8 : 0);
        v16h bfr = hcat(*(const v8h*)p2, *(const v8h*)(p2 + 16));
        a = wmma_f16(qa[kk], bfr, a);
      }
      sc[ct] = a;
    }
    // causal mask + scale + online max (row = r + 8*(lane>=16))
    #pragma unroll
    for (int r = 0; r < 8; ++r){
      const int rowabs = q0 + r + (hiHalf ? 8 : 0);
      #pragma unroll
      for (int ct = 0; ct < 2; ++ct){
        const int col = k0 + ct * 16 + l15;
        const float s = sc[ct][r] * 0.125f;          // 1/sqrt(64)
        sc[ct][r] = (col > rowabs) ? -1e30f : s;
      }
      float mx = fmaxf(sc[0][r], sc[1][r]);
      #pragma unroll
      for (int off = 8; off > 0; off >>= 1) mx = fmaxf(mx, __shfl_xor(mx, off, 32));
      const float mnew  = fmaxf(m[r], mx);
      const float alpha = __expf(m[r] - mnew);
      #pragma unroll
      for (int ct = 0; ct < 2; ++ct) sc[ct][r] = __expf(sc[ct][r] - mnew);
      #pragma unroll
      for (int nt = 0; nt < 4; ++nt) o[nt][r] *= alpha;
      lacc[r] *= alpha;
      m[r] = mnew;
    }
    // re-lay P (16x32, C layout) into LDS, reload as an A fragment
    #pragma unroll
    for (int ct = 0; ct < 2; ++ct)
      #pragma unroll
      for (int r = 0; r < 8; ++r)
        pbuf[wid][r + (hiHalf ? 8 : 0)][ct * 16 + l15] = (_Float16)sc[ct][r];
    asm volatile("" ::: "memory");   // keep DS store->load ordered in IR
    const _Float16* pp = &pbuf[wid][l15][hiHalf ? 8 : 0];
    v16h pa = hcat(*(const v8h*)pp, *(const v8h*)(pp + 16));
    // P(16x32) x V(32x64): 4 B frags from vT (contiguous along keys)
    const _Float16* vbase = vT + (((size_t)bh * DKH) << 11);
    #pragma unroll
    for (int nt = 0; nt < 4; ++nt){
      const _Float16* vp = vbase + ((size_t)(nt * 16 + l15) << 11) + k0 + (hiHalf ? 8 : 0);
      v16h vb = hcat(*(const v8h*)vp, *(const v8h*)(vp + 16));
      o[nt] = wmma_f16(pa, vb, o[nt]);
    }
    // row-sum on the matrix core: l += P x ones
    lacc = wmma_f16(pa, ones, lacc);
  }

  #pragma unroll
  for (int nt = 0; nt < 4; ++nt)
    #pragma unroll
    for (int r = 0; r < 8; ++r){
      const int rowabs = q0 + r + (hiHalf ? 8 : 0);
      ctx[(size_t)(b * SEQL + rowabs) * DM + h * DKH + nt * 16 + l15] =
          (_Float16)(o[nt][r] / lacc[r]);
    }
}

// ---------------------------------------------------------------------------
// Host launcher
// ---------------------------------------------------------------------------
extern "C" void kernel_launch(void* const* d_in, const int* in_sizes, int n_in,
                              void* d_out, int out_size, void* d_ws, size_t ws_size,
                              hipStream_t stream)
{
  (void)in_sizes; (void)n_in; (void)out_size; (void)ws_size;
  const float* x  = (const float*)d_in[0];
  const float* wq = (const float*)d_in[1];
  const float* wk = (const float*)d_in[2];
  const float* wv = (const float*)d_in[3];
  const float* wo = (const float*)d_in[4];
  const float* w1 = (const float*)d_in[5];
  const float* w2 = (const float*)d_in[6];
  const float* w3 = (const float*)d_in[7];
  const float* g1 = (const float*)d_in[8];
  const float* g2 = (const float*)d_in[9];

  char* ws = (char*)d_ws;
  auto carve = [&](size_t bytes) -> void* {
    void* p = (void*)ws; ws += (bytes + 255) & ~(size_t)255; return p;
  };
  _Float16* wq16 = (_Float16*)carve((size_t)DM * DM * 2);
  _Float16* wk16 = (_Float16*)carve((size_t)DM * DM * 2);
  _Float16* wv16 = (_Float16*)carve((size_t)DM * DM * 2);
  _Float16* wo16 = (_Float16*)carve((size_t)DM * DM * 2);
  _Float16* w116 = (_Float16*)carve((size_t)DFF * DM * 2);
  _Float16* w216 = (_Float16*)carve((size_t)DM * DFF * 2);
  _Float16* w316 = (_Float16*)carve((size_t)DFF * DM * 2);
  _Float16* xn16 = (_Float16*)carve((size_t)NTOK * DM * 2);
  _Float16* xn2  = (_Float16*)carve((size_t)NTOK * DM * 2);
  _Float16* q16  = (_Float16*)carve((size_t)NTOK * DM * 2);
  _Float16* k16  = (_Float16*)carve((size_t)NTOK * DM * 2);
  _Float16* vT16 = (_Float16*)carve((size_t)NTOK * DM * 2);
  _Float16* ctx  = (_Float16*)carve((size_t)NTOK * DM * 2);
  float*    x1   = (float*)   carve((size_t)NTOK * DM * 4);
  _Float16* h1   = (_Float16*)carve((size_t)NTOK * DFF * 2);
  _Float16* hh   = (_Float16*)carve((size_t)NTOK * DFF * 2);

  auto cvt = [&](const float* src, _Float16* dst, size_t n){
    cvt_f32_f16<<<dim3((unsigned)((n + 255) / 256)), 256, 0, stream>>>(src, dst, (int)n);
  };
  cvt(wq, wq16, (size_t)DM * DM);
  cvt(wk, wk16, (size_t)DM * DM);
  cvt(wv, wv16, (size_t)DM * DM);
  cvt(wo, wo16, (size_t)DM * DM);
  cvt(w1, w116, (size_t)DFF * DM);
  cvt(w2, w216, (size_t)DM * DFF);
  cvt(w3, w316, (size_t)DFF * DM);

  rmsnorm_kernel<<<NTOK, 256, 0, stream>>>(x, g1, xn16);

  dim3 g_proj(DM / 128, NTOK / 128);
  gemm_wmma_kernel<0><<<g_proj, 256, 0, stream>>>(xn16, wq16, q16,  nullptr, NTOK, DM, DM);
  gemm_wmma_kernel<0><<<g_proj, 256, 0, stream>>>(xn16, wk16, k16,  nullptr, NTOK, DM, DM);
  gemm_wmma_kernel<1><<<g_proj, 256, 0, stream>>>(xn16, wv16, vT16, nullptr, NTOK, DM, DM);

  rope_kernel<<<(NTOK * 512) / 256, 256, 0, stream>>>(q16);
  rope_kernel<<<(NTOK * 512) / 256, 256, 0, stream>>>(k16);

  flash_attn_kernel<<<(NB * NH * (SEQL / 16)) / 8, 256, 0, stream>>>(q16, k16, vT16, ctx);

  gemm_wmma_kernel<4><<<g_proj, 256, 0, stream>>>(ctx, wo16, x1, x, NTOK, DM, DM);

  rmsnorm_kernel<<<NTOK, 256, 0, stream>>>(x1, g2, xn2);

  dim3 g_ff1(DFF / 128, NTOK / 128);
  gemm_wmma_kernel<2><<<g_ff1, 256, 0, stream>>>(xn2, w116, h1, nullptr, NTOK, DFF, DM);
  gemm_wmma_kernel<3><<<g_ff1, 256, 0, stream>>>(xn2, w316, hh, h1,      NTOK, DFF, DM);

  dim3 g_ff2(DM / 128, NTOK / 128);
  gemm_wmma_kernel<4><<<g_ff2, 256, 0, stream>>>(hh, w216, d_out, x1, NTOK, DM, DFF);
}
